// TransformerBlock_42228118454764
// MI455X (gfx1250) — matrix operand order repaired and checked by reading, not compiled, not run on hardware
//
#include <hip/hip_runtime.h>

// ---------------------------------------------------------------------------
// CDNA5 / gfx1250 transformer block.  All GEMMs via v_wmma_f32_16x16x32_bf16.
// ---------------------------------------------------------------------------

typedef __attribute__((ext_vector_type(16))) __bf16 v16bf;
typedef __attribute__((ext_vector_type(8)))  float  v8f;

union FragU {
    uint4          q[2];    // 32 bytes
    unsigned short us[16];
    v16bf          v;
};

__device__ __forceinline__ unsigned short f32_to_bf16(float f) {
    union { float f; unsigned int u; } x; x.f = f;
    unsigned int u = x.u;
    unsigned int r = u + 0x7FFFu + ((u >> 16) & 1u);   // round-to-nearest-even
    return (unsigned short)(r >> 16);
}

// ---------------------------------------------------------------------------
// Async global->LDS staging (CDNA5 GLOBAL_LOAD_ASYNC_TO_LDS_B128, ASYNCcnt).
// Builtin signature (from hipcc diagnostic): first param is int4 AS(1)*.
// Guarded: falls back to synchronous copy if the builtins are absent.
// ---------------------------------------------------------------------------
#if defined(__HIP_DEVICE_COMPILE__) &&                                         \
    __has_builtin(__builtin_amdgcn_global_load_async_to_lds_b128) &&           \
    __has_builtin(__builtin_amdgcn_s_wait_asynccnt)
#define USE_ASYNC_LDS 1
#else
#define USE_ASYNC_LDS 0
#endif

typedef int v4i_vs __attribute__((vector_size(16)));
typedef __attribute__((address_space(1))) v4i_vs* as1_v4i;
typedef __attribute__((address_space(3))) v4i_vs* as3_v4i;

__device__ __forceinline__ void cp16_to_lds(const unsigned short* g,
                                            unsigned short* l) {
#if USE_ASYNC_LDS
    __builtin_amdgcn_global_load_async_to_lds_b128(
        (as1_v4i)(unsigned long long)(size_t)g,
        (as3_v4i)(unsigned int)(size_t)l,
        0, 0);
#else
    *(uint4*)l = *(const uint4*)g;
#endif
}

__device__ __forceinline__ void cp_lds_wait() {
#if USE_ASYNC_LDS
    __builtin_amdgcn_s_wait_asynccnt(0);
#endif
}

// ---------------------------------------------------------------------------
// Weight convert+transpose: src fp32 [K,N] -> dst bf16 [N,K]
// block (32,8), grid (K/32, N/32)
// ---------------------------------------------------------------------------
__global__ void wt_transpose_kernel(const float* __restrict__ src,
                                    unsigned short* __restrict__ dst,
                                    int K, int N) {
    __shared__ unsigned short tile[32][33];
    int tk = blockIdx.x * 32;
    int tn = blockIdx.y * 32;
    int lx = threadIdx.x;       // 0..31
    int ly = threadIdx.y;       // 0..7
    for (int i = ly; i < 32; i += 8)
        tile[i][lx] = f32_to_bf16(src[(size_t)(tk + i) * N + (tn + lx)]);
    __syncthreads();
    for (int i = ly; i < 32; i += 8)
        dst[(size_t)(tn + i) * K + (tk + lx)] = tile[lx][i];
}

// ---------------------------------------------------------------------------
// V transpose per head: v bf16 [B*T, d] -> vT bf16 [B*H, 64, T]
// block (32,8), grid (T/32, 2, B*H)
// ---------------------------------------------------------------------------
__global__ void v_transpose_kernel(const unsigned short* __restrict__ v,
                                   unsigned short* __restrict__ vT,
                                   int T, int d, int H) {
    __shared__ unsigned short tile[32][33];
    int bh = blockIdx.z;
    int b  = bh / H, h = bh % H;
    int t0 = blockIdx.x * 32;
    int j0 = blockIdx.y * 32;
    int lx = threadIdx.x, ly = threadIdx.y;
    for (int i = ly; i < 32; i += 8)
        tile[i][lx] = v[(size_t)(b * T + t0 + i) * d + h * 64 + j0 + lx];
    __syncthreads();
    for (int i = ly; i < 32; i += 8)
        vT[((size_t)bh * 64 + j0 + i) * T + t0 + lx] = tile[lx][i];
}

// ---------------------------------------------------------------------------
// LayerNorm: fp32 [M,1024] -> bf16 [M,1024].  One row per 256-thread block.
// ---------------------------------------------------------------------------
__global__ void layernorm_kernel(const float* __restrict__ x,
                                 const float* __restrict__ g,
                                 const float* __restrict__ b,
                                 unsigned short* __restrict__ out, int d) {
    __shared__ float red[256];
    int row = blockIdx.x;
    const float* xr = x + (size_t)row * d;
    float v[4];
    float s = 0.f;
    for (int i = 0; i < 4; i++) { v[i] = xr[threadIdx.x + 256 * i]; s += v[i]; }
    red[threadIdx.x] = s; __syncthreads();
    for (int off = 128; off > 0; off >>= 1) {
        if (threadIdx.x < off) red[threadIdx.x] += red[threadIdx.x + off];
        __syncthreads();
    }
    float mu = red[0] / (float)d;
    __syncthreads();
    float s2 = 0.f;
    for (int i = 0; i < 4; i++) { float t = v[i] - mu; s2 += t * t; }
    red[threadIdx.x] = s2; __syncthreads();
    for (int off = 128; off > 0; off >>= 1) {
        if (threadIdx.x < off) red[threadIdx.x] += red[threadIdx.x + off];
        __syncthreads();
    }
    float rstd = rsqrtf(red[0] / (float)d + 1e-5f);
    for (int i = 0; i < 4; i++) {
        int c = threadIdx.x + 256 * i;
        float y = (v[i] - mu) * rstd * g[c] + b[c];
        out[(size_t)row * d + c] = f32_to_bf16(y);
    }
}

// ---------------------------------------------------------------------------
// WMMA GEMM: C[M,N] = epilogue(A_bf16[M,K] @ Wt_bf16[N,K]^T + bias[N])
// block = 128 threads (4 waves), tile 128x64, K-step 32.
// Wave w computes rows w*32..w*32+31 x all 64 cols: 8 WMMAs per K-step.
// ---------------------------------------------------------------------------
#define GEMM_BF16 0
#define GEMM_RES  1
#define GEMM_GELU 2

template <int MODE>
__global__ __launch_bounds__(128)
void gemm_wmma_kernel(const unsigned short* __restrict__ A,
                      const unsigned short* __restrict__ Wt,
                      const float* __restrict__ bias,
                      const float* __restrict__ res,
                      void* __restrict__ out,
                      int M, int N, int K) {
    __shared__ __align__(16) unsigned short As[128 * 32];
    __shared__ __align__(16) unsigned short Bs[64 * 32];

    int bn = blockIdx.x * 64;
    int bm = blockIdx.y * 128;
    int tid  = threadIdx.x;
    int lane = tid & 31;
    int wave = tid >> 5;
    int lh   = (lane < 16) ? 0 : 1;
    int ln15 = lane & 15;

    v8f acc[2][4];
    for (int mi = 0; mi < 2; mi++)
        for (int i = 0; i < 4; i++)
            for (int j = 0; j < 8; j++) acc[mi][i][j] = 0.f;

    for (int kb = 0; kb < K; kb += 32) {
        // stage 128x32 A tile (512 uint4) and 64x32 Wt tile (256 uint4)
        #pragma unroll
        for (int it = 0; it < 4; it++) {
            int idx = tid + it * 128;           // 0..511
            int r   = idx >> 2;                 // 0..127
            int c   = (idx & 3) * 8;            // 0,8,16,24
            cp16_to_lds(A + (size_t)(bm + r) * K + kb + c, As + r * 32 + c);
        }
        #pragma unroll
        for (int it = 0; it < 2; it++) {
            int idx = tid + it * 128;           // 0..255
            int r   = idx >> 2;                 // 0..63
            int c   = (idx & 3) * 8;
            cp16_to_lds(Wt + (size_t)(bn + r) * K + kb + c, Bs + r * 32 + c);
        }
        if (kb + 32 < K) {
            __builtin_prefetch(A  + (size_t)(bm + tid) * K + kb + 32, 0, 1);
            __builtin_prefetch(Wt + (size_t)(bn + (tid >> 1)) * K + kb + 32, 0, 1);
        }
        cp_lds_wait();
        __syncthreads();

        // A fragments: rows wave*32 + {0..15, 16..31}
        FragU af[2];
        #pragma unroll
        for (int mi = 0; mi < 2; mi++) {
            const unsigned short* ap =
                As + (wave * 32 + mi * 16 + ln15) * 32 + lh * 8;
            af[mi].q[0] = *(const uint4*)(ap);
            af[mi].q[1] = *(const uint4*)(ap + 16);
        }
        #pragma unroll
        for (int nt = 0; nt < 4; nt++) {
            FragU bf;
            const unsigned short* bp = Bs + (nt * 16 + ln15) * 32 + lh * 8;
            bf.q[0] = *(const uint4*)(bp);
            bf.q[1] = *(const uint4*)(bp + 16);
            #pragma unroll
            for (int mi = 0; mi < 2; mi++)
                acc[mi][nt] = __builtin_amdgcn_wmma_f32_16x16x32_bf16(
                    false, af[mi].v, false, bf.v, (short)0, acc[mi][nt],
                    false, false);
        }
        __syncthreads();
    }

    // epilogue; C layout: lane ln15 = column, VGPR r = row (r + lh*8)
    #pragma unroll
    for (int nt = 0; nt < 4; nt++) {
        int colg = bn + nt * 16 + ln15;
        float bv = bias[colg];
        #pragma unroll
        for (int mi = 0; mi < 2; mi++) {
            int m0 = wave * 32 + mi * 16 + lh * 8;
            #pragma unroll
            for (int r = 0; r < 8; r++) {
                int rowg = bm + m0 + r;
                size_t off = (size_t)rowg * N + colg;
                float val = acc[mi][nt][r] + bv;
                if (MODE == GEMM_BF16) {
                    ((unsigned short*)out)[off] = f32_to_bf16(val);
                } else if (MODE == GEMM_RES) {
                    ((float*)out)[off] = res[off] + val;
                } else {  // exact GELU
                    float g = 0.5f * val *
                              (1.0f + erff(val * 0.70710678118654752f));
                    ((unsigned short*)out)[off] = f32_to_bf16(g);
                }
            }
        }
    }
}

// ---------------------------------------------------------------------------
// Causal flash attention.  One wave per 16-row Q tile; key blocks of 32.
// Computes S^T = K*Q^T so the P^T C-layout is directly the A-fragment of P*V.
// q,k: bf16 [B*T, d] (head h at column h*64);  vT: bf16 [B*H, 64, T]
// o: bf16 [B*T, d]
// ---------------------------------------------------------------------------
__global__ __launch_bounds__(128)
void attention_kernel(const unsigned short* __restrict__ q,
                      const unsigned short* __restrict__ k,
                      const unsigned short* __restrict__ vT,
                      unsigned short* __restrict__ o,
                      int B, int T, int H) {
    const int d   = H * 64;
    int lane = threadIdx.x & 31;
    int wgl  = threadIdx.x >> 5;
    int lh   = (lane < 16) ? 0 : 1;
    int ln15 = lane & 15;

    int wg    = blockIdx.x * 4 + wgl;      // global wave index
    int ntile = T / 16;
    int bh    = wg / ntile;
    int qt    = wg % ntile;
    int b     = bh / H, h = bh % H;
    int qbase = qt * 16;

    // Q^T B-fragments (lane = q column), loaded once
    const unsigned short* qrow = q + (size_t)(b * T + qbase + ln15) * d + h * 64;
    FragU qf[2];
    #pragma unroll
    for (int c = 0; c < 2; c++) {
        const unsigned short* p = qrow + c * 32 + lh * 8;
        qf[c].q[0] = *(const uint4*)(p);
        qf[c].q[1] = *(const uint4*)(p + 16);
    }

    v8f acc[4];
    for (int i = 0; i < 4; i++)
        for (int j = 0; j < 8; j++) acc[i][j] = 0.f;
    float mrow = -1e30f;   // running max for q-row = ln15
    float lrow = 0.f;      // running sum
    const float sc = 0.125f;   // 1/sqrt(64)
    int qg = qbase + ln15;

    int nkb = (qbase + 47) >> 5;           // number of 32-key blocks (causal)
    for (int kbidx = 0; kbidx < nkb; kbidx++) {
        int key0 = kbidx * 32;

        // S^T (two 16-key tiles), M=key, N=q
        v8f st[2];
        for (int i = 0; i < 2; i++)
            for (int j = 0; j < 8; j++) st[i][j] = 0.f;
        #pragma unroll
        for (int kt = 0; kt < 2; kt++) {
            const unsigned short* krow =
                k + (size_t)(b * T + key0 + kt * 16 + ln15) * d + h * 64;
            #pragma unroll
            for (int c = 0; c < 2; c++) {
                FragU kf;
                const unsigned short* p = krow + c * 32 + lh * 8;
                kf.q[0] = *(const uint4*)(p);
                kf.q[1] = *(const uint4*)(p + 16);
                st[kt] = __builtin_amdgcn_wmma_f32_16x16x32_bf16(
                    false, kf.v, false, qf[c].v, (short)0, st[kt], false, false);
            }
        }

        // scale + causal mask + block max (per lane: fixed q, 16 keys)
        float bmax = -1e30f;
        #pragma unroll
        for (int kt = 0; kt < 2; kt++)
            #pragma unroll
            for (int r = 0; r < 8; r++) {
                int kg = key0 + kt * 16 + lh * 8 + r;
                float s = (kg <= qg) ? st[kt][r] * sc : -1e30f;
                st[kt][r] = s;
                bmax = fmaxf(bmax, s);
            }
        bmax = fmaxf(bmax, __shfl_xor(bmax, 16, 32));   // merge lane halves
        float mnew  = fmaxf(mrow, bmax);
        float scale = __expf(mrow - mnew);

        // P^T = exp(S^T - mnew); pack straight into an A-fragment (bf16)
        FragU pf;
        float psum = 0.f;
        #pragma unroll
        for (int kt = 0; kt < 2; kt++)
            #pragma unroll
            for (int r = 0; r < 8; r++) {
                float p = __expf(st[kt][r] - mnew);
                psum += p;
                pf.us[kt * 8 + r] = f32_to_bf16(p);
            }
        lrow = lrow * scale + (psum + __shfl_xor(psum, 16, 32));
        mrow = mnew;

        // rescale accumulators: acc VGPR r holds q-row (r + lh*8); that row's
        // scale lives in lane (r + lh*8) of the 16-lane group
        #pragma unroll
        for (int r = 0; r < 8; r++) {
            float srow = __shfl(scale, r + lh * 8, 16);
            #pragma unroll
            for (int nt = 0; nt < 4; nt++) acc[nt][r] *= srow;
        }

        // O += P @ V  (B-fragments from vT, lane = hd column)
        #pragma unroll
        for (int nt = 0; nt < 4; nt++) {
            FragU vf;
            const unsigned short* vp =
                vT + ((size_t)bh * 64 + nt * 16 + ln15) * T + key0 + lh * 8;
            vf.q[0] = *(const uint4*)(vp);
            vf.q[1] = *(const uint4*)(vp + 16);
            acc[nt] = __builtin_amdgcn_wmma_f32_16x16x32_bf16(
                false, pf.v, false, vf.v, (short)0, acc[nt], false, false);
        }
    }

    // normalize and store bf16 output
    float inv = 1.0f / lrow;
    #pragma unroll
    for (int r = 0; r < 8; r++) {
        float sinv = __shfl(inv, r + lh * 8, 16);
        int rowg = b * T + qbase + lh * 8 + r;
        #pragma unroll
        for (int nt = 0; nt < 4; nt++)
            o[(size_t)rowg * d + h * 64 + nt * 16 + ln15] =
                f32_to_bf16(acc[nt][r] * sinv);
    }
}

// ---------------------------------------------------------------------------
// Host launch
// ---------------------------------------------------------------------------
extern "C" void kernel_launch(void* const* d_in, const int* in_sizes, int n_in,
                              void* d_out, int out_size, void* d_ws, size_t ws_size,
                              hipStream_t stream) {
    (void)in_sizes; (void)n_in; (void)out_size; (void)ws_size;
    const int B = 4, T = 2048, D = 1024, H = 16, DFF = 4096;
    const int M = B * T;

    const float* x   = (const float*)d_in[0];
    const float* Wq  = (const float*)d_in[1];
    const float* bq  = (const float*)d_in[2];
    const float* Wk  = (const float*)d_in[3];
    const float* bk  = (const float*)d_in[4];
    const float* Wv  = (const float*)d_in[5];
    const float* bv  = (const float*)d_in[6];
    const float* Wp  = (const float*)d_in[7];
    const float* bp  = (const float*)d_in[8];
    const float* g1  = (const float*)d_in[9];
    const float* b1  = (const float*)d_in[10];
    const float* g2  = (const float*)d_in[11];
    const float* b2  = (const float*)d_in[12];
    const float* W1  = (const float*)d_in[13];
    const float* bf1 = (const float*)d_in[14];
    const float* W2  = (const float*)d_in[15];
    const float* bf2 = (const float*)d_in[16];
    float* out = (float*)d_out;

    char* ws = (char*)d_ws;
    size_t off = 0;
    auto alloc = [&](size_t bytes) -> char* {
        char* p = ws + off;
        off += (bytes + 255) & ~(size_t)255;
        return p;
    };
    unsigned short* xn  = (unsigned short*)alloc((size_t)M * D * 2);   // LN1 out; reused for LN2 out
    unsigned short* Wqt = (unsigned short*)alloc((size_t)D * D * 2);
    unsigned short* Wkt = (unsigned short*)alloc((size_t)D * D * 2);
    unsigned short* Wvt = (unsigned short*)alloc((size_t)D * D * 2);
    unsigned short* Wpt = (unsigned short*)alloc((size_t)D * D * 2);
    unsigned short* W1t = (unsigned short*)alloc((size_t)D * DFF * 2);
    unsigned short* W2t = (unsigned short*)alloc((size_t)D * DFF * 2);
    unsigned short* qb  = (unsigned short*)alloc((size_t)M * D * 2);
    unsigned short* kb  = (unsigned short*)alloc((size_t)M * D * 2);
    unsigned short* vb  = (unsigned short*)alloc((size_t)M * D * 2);   // reused as attn out
    unsigned short* vTb = (unsigned short*)alloc((size_t)M * D * 2);
    float*          ar  = (float*)alloc((size_t)M * D * 4);
    unsigned short* hb  = (unsigned short*)alloc((size_t)M * DFF * 2);

    dim3 tb(32, 8, 1);
    // weight convert + transpose to bf16 [N,K]
    wt_transpose_kernel<<<dim3(D / 32, D / 32), tb, 0, stream>>>(Wq, Wqt, D, D);
    wt_transpose_kernel<<<dim3(D / 32, D / 32), tb, 0, stream>>>(Wk, Wkt, D, D);
    wt_transpose_kernel<<<dim3(D / 32, D / 32), tb, 0, stream>>>(Wv, Wvt, D, D);
    wt_transpose_kernel<<<dim3(D / 32, D / 32), tb, 0, stream>>>(Wp, Wpt, D, D);
    wt_transpose_kernel<<<dim3(D / 32, DFF / 32), tb, 0, stream>>>(W1, W1t, D, DFF);
    wt_transpose_kernel<<<dim3(DFF / 32, D / 32), tb, 0, stream>>>(W2, W2t, DFF, D);

    // LN1
    layernorm_kernel<<<dim3(M), dim3(256), 0, stream>>>(x, g1, b1, xn, D);

    // QKV projections
    gemm_wmma_kernel<GEMM_BF16><<<dim3(D / 64, M / 128), dim3(128), 0, stream>>>(
        xn, Wqt, bq, nullptr, qb, M, D, D);
    gemm_wmma_kernel<GEMM_BF16><<<dim3(D / 64, M / 128), dim3(128), 0, stream>>>(
        xn, Wkt, bk, nullptr, kb, M, D, D);
    gemm_wmma_kernel<GEMM_BF16><<<dim3(D / 64, M / 128), dim3(128), 0, stream>>>(
        xn, Wvt, bv, nullptr, vb, M, D, D);

    // V transpose per head
    v_transpose_kernel<<<dim3(T / 32, 2, B * H), tb, 0, stream>>>(vb, vTb, T, D, H);

    // causal flash attention (attn output reuses vb)
    unsigned short* ab = vb;
    attention_kernel<<<dim3(B * H * (T / 16) / 4), dim3(128), 0, stream>>>(
        qb, kb, vTb, ab, B, T, H);

    // output projection + residual: ar = x + a @ Wp + bp
    gemm_wmma_kernel<GEMM_RES><<<dim3(D / 64, M / 128), dim3(128), 0, stream>>>(
        ab, Wpt, bp, x, ar, M, D, D);

    // LN2 (into xn)
    layernorm_kernel<<<dim3(M), dim3(256), 0, stream>>>(ar, g2, b2, xn, D);

    // FFN1 + GELU
    gemm_wmma_kernel<GEMM_GELU><<<dim3(DFF / 64, M / 128), dim3(128), 0, stream>>>(
        xn, W1t, bf1, nullptr, hb, M, DFF, D);

    // FFN2 + residual: out = ar + h @ W2 + bf2
    gemm_wmma_kernel<GEMM_RES><<<dim3(D / 64, M / 128), dim3(128), 0, stream>>>(
        hb, W2t, bf2, ar, out, M, D, DFF);
}
